// TAGEncoder_27023934227225
// MI455X (gfx1250) — matrix operand order repaired and checked by reading, not compiled, not run on hardware
//
#include <hip/hip_runtime.h>
#include <hip/hip_bf16.h>

// ---------------------------------------------------------------------------
// TAGConv (K=3) x2 + leaky_relu + log_softmax for MI455X (gfx1250, wave32).
// GEMMs use v_wmma_f32_16x16x32_bf16 with vectorized LDS fragment traffic
// (ds_load_b128 / ds_store_b64); sparse hops use L2-resident float atomics.
// ---------------------------------------------------------------------------

#define GN      20000      // nodes
#define GF_IN   1000       // input features
#define GHID    512
#define GLAT    128
#define GE      300000     // edges
#define KHOPS   3

typedef __attribute__((ext_vector_type(16))) __bf16 v16bf;
typedef __attribute__((ext_vector_type(8)))  __bf16 v8bf;
typedef __attribute__((ext_vector_type(4)))  __bf16 v4bf;
typedef __attribute__((ext_vector_type(8)))  float  v8f;

// ------------------------------ small kernels ------------------------------

__global__ void zero_kernel(float* __restrict__ p, long n) {
  long i = (long)blockIdx.x * blockDim.x + threadIdx.x;
  if (i < n) p[i] = 0.0f;
}

__global__ void degree_kernel(const long long* __restrict__ dst,
                              float* __restrict__ deg, int E) {
  int e = blockIdx.x * blockDim.x + threadIdx.x;
  if (e < E) atomicAdd(&deg[(int)dst[e]], 1.0f);
}

__global__ void dinv_kernel(float* __restrict__ deg, int n) {
  int i = blockIdx.x * blockDim.x + threadIdx.x;
  if (i < n) {
    float d = deg[i];
    deg[i] = (d > 0.0f) ? rsqrtf(fmaxf(d, 1.0f)) : 0.0f;
  }
}

__global__ void norm_kernel(const long long* __restrict__ src,
                            const long long* __restrict__ dst,
                            const float* __restrict__ dinv,
                            float* __restrict__ norm, int E) {
  int e = blockIdx.x * blockDim.x + threadIdx.x;
  if (e < E) norm[e] = dinv[(int)src[e]] * dinv[(int)dst[e]];
}

// one hop of normalized propagation: out[dst] += norm[e] * h[src]
// F is a multiple of 4 (1000 / 512) -> float4 gather, scalar f32 atomics.
__global__ void propagate_kernel(const float* __restrict__ h,
                                 float* __restrict__ out,
                                 const long long* __restrict__ src,
                                 const long long* __restrict__ dst,
                                 const float* __restrict__ norm, int F4) {
  int e = blockIdx.x;
  long long s = src[e], d = dst[e];
  float w = norm[e];
  const float4* hs = (const float4*)(h + (size_t)s * (F4 * 4));
  float* od = out + (size_t)d * (F4 * 4);
  for (int f = threadIdx.x; f < F4; f += blockDim.x) {
    float4 v = hs[f];
    atomicAdd(&od[f * 4 + 0], w * v.x);
    atomicAdd(&od[f * 4 + 1], w * v.y);
    atomicAdd(&od[f * 4 + 2], w * v.z);
    atomicAdd(&od[f * 4 + 3], w * v.w);
  }
}

__global__ void leaky_relu_kernel(float* __restrict__ p, long n) {
  long i = (long)blockIdx.x * blockDim.x + threadIdx.x;
  if (i < n) {
    float v = p[i];
    p[i] = (v > 0.0f) ? v : 0.02f * v;
  }
}

// ----------------------------- WMMA GEMM -----------------------------------
// C[M,Nout] (+)= A[M,K] @ B[K,Nout] (+ bias). bf16 inputs, f32 accumulate.
// Block tile 128x64, 128 threads (4 waves); each wave owns a 32x64 strip
// (2 M-substrips x 4 N-tiles = 8 WMMAs per 32-wide K chunk).

#define TM 128
#define TN 64
#define TK 32

__global__ __launch_bounds__(128) void gemm_wmma_bf16(
    const float* __restrict__ A, const float* __restrict__ B,
    float* __restrict__ C, const float* __restrict__ bias,
    int M, int K, int Nout, int accumulate) {
  __shared__ __align__(16) __bf16 sA[TM * TK];   // [m][k] row-major, 8 KB
  __shared__ __align__(16) __bf16 sBt[TN * TK];  // [n][k] transposed, 4 KB

  const int tid  = threadIdx.x;
  const int wave = tid >> 5;
  const int lane = tid & 31;
  const int lr   = lane & 15;
  const int hsel = (lane < 16) ? 0 : 1;   // lane half select
  const int m0   = blockIdx.x * TM;
  const int n0   = blockIdx.y * TN;

  v8f acc[2][4] = {};   // [M substrip][N tile]

  for (int k0 = 0; k0 < K; k0 += TK) {
    // prefetch next A chunk (one row per thread) -> global_prefetch_b8
    if (k0 + TK < K && m0 + tid < M)
      __builtin_prefetch(&A[(size_t)(m0 + tid) * K + k0 + TK], 0, 1);

    // stage A: quads of 4 consecutive k; one ds_store_b64 per quad
    for (int t = tid; t < TM * TK / 4; t += 128) {
      int r  = t >> 3;            // TK/4 = 8 quads per row
      int c4 = (t & 7) * 4;
      int gm = m0 + r;
      v4bf pk;
#pragma unroll
      for (int u = 0; u < 4; ++u) {
        int gk = k0 + c4 + u;
        float v = (gm < M && gk < K) ? A[(size_t)gm * K + gk] : 0.0f;
        pk[u] = (__bf16)v;
      }
      *(v4bf*)&sA[r * TK + c4] = pk;
    }
    // stage B transposed: threads 0..63 share k -> coalesced global reads
    // along n; contiguous ds_store_b64 along k.
    for (int t = tid; t < TN * TK / 4; t += 128) {
      int n  = t & 63;
      int k4 = (t >> 6) * 4;
      v4bf pk;
#pragma unroll
      for (int u = 0; u < 4; ++u) {
        int gk = k0 + k4 + u;
        float v = (gk < K) ? B[(size_t)gk * Nout + (n0 + n)] : 0.0f;
        pk[u] = (__bf16)v;
      }
      *(v4bf*)&sBt[n * TK + k4] = pk;
    }
    __syncthreads();

    // A fragments (2 substrips): lanes 0-15 K{0..7,16..23}, 16-31 K{8..15,24..31}
    v16bf afrag[2];
#pragma unroll
    for (int s = 0; s < 2; ++s) {
      int m  = wave * 32 + s * 16 + lr;
      int kb = hsel * 8;
      v8bf lo = *(const v8bf*)&sA[m * TK + kb];
      v8bf hi = *(const v8bf*)&sA[m * TK + kb + 16];
#pragma unroll
      for (int i = 0; i < 8; ++i) { afrag[s][i] = lo[i]; afrag[s][8 + i] = hi[i]; }
    }
#pragma unroll
    for (int j = 0; j < 4; ++j) {
      // B fragment: col = lr; lanes 0-15 K{0..7,16..23}, 16-31 K{8..15,24..31}
      int n  = j * 16 + lr;
      int kb = hsel * 8;
      v8bf lo = *(const v8bf*)&sBt[n * TK + kb];
      v8bf hi = *(const v8bf*)&sBt[n * TK + kb + 16];
      v16bf bfrag;
#pragma unroll
      for (int i = 0; i < 8; ++i) { bfrag[i] = lo[i]; bfrag[8 + i] = hi[i]; }
      acc[0][j] = __builtin_amdgcn_wmma_f32_16x16x32_bf16(
          false, afrag[0], false, bfrag, (short)0, acc[0][j], false, false);
      acc[1][j] = __builtin_amdgcn_wmma_f32_16x16x32_bf16(
          false, afrag[1], false, bfrag, (short)0, acc[1][j], false, false);
    }
    __syncthreads();
  }

  // C/D layout: VGPR r -> row (substrip base + r + hsel*8), col = lr (+16j)
#pragma unroll
  for (int s = 0; s < 2; ++s) {
#pragma unroll
    for (int j = 0; j < 4; ++j) {
      int gn = n0 + j * 16 + lr;
#pragma unroll
      for (int r = 0; r < 8; ++r) {
        int gm = m0 + wave * 32 + s * 16 + r + hsel * 8;
        if (gm < M) {
          size_t off = (size_t)gm * Nout + gn;
          float v = acc[s][j][r];
          if (accumulate) C[off] += v;
          else            C[off] = v + (bias ? bias[gn] : 0.0f);
        }
      }
    }
  }
}

// --------------------------- row log-softmax -------------------------------
// one 128-thread block per row of 128 cols, in place; includes reference +1e-6

__global__ __launch_bounds__(128) void log_softmax_kernel(float* __restrict__ p) {
  __shared__ float red[4];
  int row = blockIdx.x, tid = threadIdx.x;
  float v = p[(size_t)row * GLAT + tid] + 1e-6f;

  float m = v;
  for (int o = 16; o > 0; o >>= 1) m = fmaxf(m, __shfl_xor(m, o, 32));
  if ((tid & 31) == 0) red[tid >> 5] = m;
  __syncthreads();
  float mx = fmaxf(fmaxf(red[0], red[1]), fmaxf(red[2], red[3]));

  float e = __expf(v - mx);
  float s = e;
  for (int o = 16; o > 0; o >>= 1) s += __shfl_xor(s, o, 32);
  __syncthreads();
  if ((tid & 31) == 0) red[tid >> 5] = s;
  __syncthreads();
  float sum = red[0] + red[1] + red[2] + red[3];

  p[(size_t)row * GLAT + tid] = v - mx - __logf(sum);
}

// ------------------------------- launcher ----------------------------------

static inline void zero_async(float* p, long n, hipStream_t st) {
  zero_kernel<<<(unsigned)((n + 255) / 256), 256, 0, st>>>(p, n);
}

extern "C" void kernel_launch(void* const* d_in, const int* in_sizes, int n_in,
                              void* d_out, int out_size, void* d_ws, size_t ws_size,
                              hipStream_t stream) {
  const float*     x   = (const float*)d_in[0];
  const long long* ei  = (const long long*)d_in[1];   // [2, E] int64
  const float*     W1  = (const float*)d_in[2];        // [4, F_IN, HID]
  const float*     b1  = (const float*)d_in[3];
  const float*     W2  = (const float*)d_in[4];        // [4, HID, LAT]
  const float*     b2  = (const float*)d_in[5];
  float*           out = (float*)d_out;                // [N, LAT]

  const long long* src = ei;
  const long long* dst = ei + GE;

  // workspace carve-up (floats)
  float* ws   = (float*)d_ws;
  float* deg  = ws;                                    // N  (becomes dinv)
  float* norm = deg + GN;                              // E
  float* hA   = norm + GE;                             // N*F_IN
  float* hB   = hA + (size_t)GN * GF_IN;               // N*F_IN
  float* out1 = hB + (size_t)GN * GF_IN;               // N*HID

  // ---- gcn_norm ----
  zero_async(deg, GN, stream);
  degree_kernel<<<(GE + 255) / 256, 256, 0, stream>>>(dst, deg, GE);
  dinv_kernel<<<(GN + 255) / 256, 256, 0, stream>>>(deg, GN);
  norm_kernel<<<(GE + 255) / 256, 256, 0, stream>>>(src, dst, deg, norm, GE);

  dim3 blk(128);
  dim3 g1((GN + TM - 1) / TM, GHID / TN);   // layer-1 GEMM grid
  dim3 g2((GN + TM - 1) / TM, GLAT / TN);   // layer-2 GEMM grid

  // ---- layer 1: out1 = x@W1[0] + b1 ; hops accumulate ----
  gemm_wmma_bf16<<<g1, blk, 0, stream>>>(x, W1, out1, b1, GN, GF_IN, GHID, 0);
  {
    const float* hprev = x;
    float* bufs[2] = {hA, hB};
    for (int k = 1; k <= KHOPS; ++k) {
      float* hn = bufs[(k - 1) & 1];
      zero_async(hn, (long)GN * GF_IN, stream);
      propagate_kernel<<<GE, 128, 0, stream>>>(hprev, hn, src, dst, norm,
                                               GF_IN / 4);
      gemm_wmma_bf16<<<g1, blk, 0, stream>>>(hn, W1 + (size_t)k * GF_IN * GHID,
                                             out1, nullptr, GN, GF_IN, GHID, 1);
      hprev = hn;
    }
  }
  leaky_relu_kernel<<<(unsigned)(((long)GN * GHID + 255) / 256), 256, 0, stream>>>(
      out1, (long)GN * GHID);

  // ---- layer 2: out = out1@W2[0] + b2 ; hops accumulate (into d_out) ----
  gemm_wmma_bf16<<<g2, blk, 0, stream>>>(out1, W2, out, b2, GN, GHID, GLAT, 0);
  {
    const float* hprev = out1;
    float* bufs[2] = {hA, hB};
    for (int k = 1; k <= KHOPS; ++k) {
      float* hn = bufs[(k - 1) & 1];
      zero_async(hn, (long)GN * GHID, stream);
      propagate_kernel<<<GE, 128, 0, stream>>>(hprev, hn, src, dst, norm,
                                               GHID / 4);
      gemm_wmma_bf16<<<g2, blk, 0, stream>>>(hn, W2 + (size_t)k * GHID * GLAT,
                                             out, nullptr, GN, GHID, GLAT, 1);
      hprev = hn;
    }
  }

  // ---- log_softmax(h + 1e-6) in place on d_out ----
  log_softmax_kernel<<<GN, 128, 0, stream>>>(out);
}
